// StyleLoss_3676492005883
// MI455X (gfx1250) — compile-verified, matrix operand order checked by reading, not run on hardware
//
#include <hip/hip_runtime.h>
#include <hip/hip_bf16.h>

// ---------------- CDNA5 vector types ----------------
typedef __attribute__((ext_vector_type(16))) _Float16 v16h;
typedef __attribute__((ext_vector_type(8)))  _Float16 v8h;
typedef __attribute__((ext_vector_type(8)))  float    v8f;
typedef __attribute__((ext_vector_type(4)))  int      v4i;

// ---------------- gfx1250 async global->LDS path (guarded) ----------------
#if defined(__gfx1250__) && __has_builtin(__builtin_amdgcn_global_load_async_to_lds_b128)
#define HAVE_ASYNC_LDS 1
#else
#define HAVE_ASYNC_LDS 0
#endif

#if HAVE_ASYNC_LDS
#if __has_builtin(__builtin_amdgcn_s_wait_asynccnt)
#define WAIT_ASYNC(n) __builtin_amdgcn_s_wait_asynccnt(n)
#else
#define WAIT_ASYNC(n) asm volatile("s_wait_asynccnt %0" ::"n"(n) : "memory")
#endif
#else
#define WAIT_ASYNC(n)
#endif

// ---------------- problem constants ----------------
constexpr int C = 192, H = 128, W = 128;
constexpr int NP = 63;            // patch positions per axis (stride 2, patch 3)
constexpr int NQ = NP * NP;       // 3969 synth patches
constexpr int NS = 2945;          // kept style patches (3969 - 32*32 interior)
constexpr int D  = C * 9;         // 1728 patch dim

constexpr int S_PAD = 3072;
constexpr int Q_PAD = 4096;
constexpr int S_CHUNKS = 8;
constexpr int S_SPAN = S_PAD / S_CHUNKS;  // 384
constexpr int BS = 64, BQ = 128, BK = 64; // GEMM tiles (BK = 2 WMMA K-steps)
constexpr int KT = D / BK;                // 27 K-tiles
constexpr int LDSW = BK + 8;              // 72 f16 row stride (bank stagger)
constexpr int NBUF = 3;                   // triple buffer -> 1 barrier/tile

// ---------------- workspace layout (bytes) ----------------
constexpr size_t OFF_STY  = 0;                                        // f16 [S_PAD][D]
constexpr size_t SZ_STY   = (size_t)S_PAD * D * 2;
constexpr size_t OFF_SYN  = OFF_STY + SZ_STY;                         // f16 [Q_PAD][D]
constexpr size_t SZ_SYN   = (size_t)Q_PAD * D * 2;
constexpr size_t OFF_INVN = OFF_SYN + SZ_SYN;                         // f32 [S_PAD]
constexpr size_t OFF_SRC  = OFF_INVN + (size_t)S_PAD * 4;             // i32 [S_PAD]
constexpr size_t OFF_PBST = OFF_SRC  + (size_t)S_PAD * 4;             // f32 [S_CHUNKS][Q_PAD]
constexpr size_t OFF_PIDX = OFF_PBST + (size_t)S_CHUNKS * Q_PAD * 4;  // i32 [S_CHUNKS][Q_PAD]

// ---------------- 16B mover: async-to-LDS on gfx1250, sync fallback ----------------
__device__ __forceinline__ void cp16(const _Float16* g, _Float16* l) {
#if HAVE_ASYNC_LDS
  __builtin_amdgcn_global_load_async_to_lds_b128(
      (__attribute__((address_space(1))) v4i*)g,
      (__attribute__((address_space(3))) v4i*)l, 0, 0);
#else
  *(v8h*)l = *(const v8h*)g;
#endif
}

// ---------------- stage 0: style keep-mask scatter + zero loss ----------------
__global__ void k0_setup(int* __restrict__ styleSrc, float* __restrict__ out) {
  int p = blockIdx.x * 256 + threadIdx.x;
  if (p == 0) out[0] = 0.0f;
  if (p >= NQ) return;
  int i = p / NP, j = p % NP;
  bool ii = (i >= 16 && i <= 47);
  bool jj = (j >= 16 && j <= 47);
  if (ii && jj) return;                       // interior -> dropped
  int rowsBefore = min(max(i - 16, 0), 32);
  int before = rowsBefore * 32 + (ii ? min(max(j - 16, 0), 32) : 0);
  styleSrc[p - before] = p;
}

// ---------------- stage 1a: style flats (f16) + inverse norms (f32) ----------------
__global__ void k1_style(const float* __restrict__ target, const int* __restrict__ styleSrc,
                         _Float16* __restrict__ sty, float* __restrict__ invNorm) {
  __shared__ float red[8];
  const int s = blockIdx.x, tid = threadIdx.x;
  float sum = 0.0f;
  if (s < NS) {
    int p = styleSrc[s];
    int pi = (p / NP) * 2, pj = (p % NP) * 2;
    for (int d = tid; d < D; d += 256) {
      int c = d / 9, r = d % 9;
      float v = target[c * (H * W) + (pi + r / 3) * W + (pj + r % 3)];
      sty[(size_t)s * D + d] = (_Float16)v;
      sum += v * v;
    }
  } else {
    for (int d = tid; d < D; d += 256) sty[(size_t)s * D + d] = (_Float16)0.0f;
  }
  for (int off = 16; off; off >>= 1) sum += __shfl_down(sum, off, 32);
  if ((tid & 31) == 0) red[tid >> 5] = sum;
  __syncthreads();
  if (tid == 0) {
    float t = 0.0f;
    #pragma unroll
    for (int w = 0; w < 8; ++w) t += red[w];
    invNorm[s] = (s < NS && t > 0.0f) ? rsqrtf(t) : 0.0f;
  }
}

// ---------------- stage 1b: synth flats (f16) ----------------
__global__ void k1_synth(const float* __restrict__ img, _Float16* __restrict__ syn) {
  const int q = blockIdx.x, tid = threadIdx.x;
  if (q < NQ) {
    int pi = (q / NP) * 2, pj = (q % NP) * 2;
    for (int d = tid; d < D; d += 256) {
      int c = d / 9, r = d % 9;
      syn[(size_t)q * D + d] = (_Float16)img[c * (H * W) + (pi + r / 3) * W + (pj + r % 3)];
    }
  } else {
    for (int d = tid; d < D; d += 256) syn[(size_t)q * D + d] = (_Float16)0.0f;
  }
}

// ---------------- stage 2: WMMA GEMM + fused argmax ----------------
__global__ __launch_bounds__(256)
void k2_gemm_argmax(const _Float16* __restrict__ sty, const _Float16* __restrict__ syn,
                    const float* __restrict__ invNorm,
                    float* __restrict__ pBest, int* __restrict__ pIdx) {
  __shared__ __align__(16) _Float16 ldsA[NBUF][BS * LDSW];   // style  64 x 64
  __shared__ __align__(16) _Float16 ldsB[NBUF][BQ * LDSW];   // synth 128 x 64
  __shared__ float invLds[BS];
  __shared__ float bestLds[2][BQ];
  __shared__ int   idxLds[2][BQ];

  const int tid  = threadIdx.x;
  const int lane = tid & 31;
  const int wave = tid >> 5;       // 0..7
  const int wm   = wave >> 2;      // s-split 0..1
  const int wn   = wave & 3;       // q-split 0..3
  const int g    = lane >> 4;      // lane half -> K-range select (ISA layout)
  const int r    = lane & 15;

  const int qBase      = blockIdx.x * BQ;
  const int sChunkBase = blockIdx.y * S_SPAN;

  // cooperative tile-copy mapping (256 threads)
  const int arow = tid >> 2, acol = (tid & 3) * 16;  // A: 64 x 64, 2x b128/thread
  const int brow = tid >> 1, bcol = (tid & 1) * 32;  // B: 128 x 64, 4x b128/thread
  const _Float16* gB = syn + (size_t)(qBase + brow) * D + bcol;  // + k
  _Float16* lA0 = &ldsA[0][arow * LDSW + acol];
  _Float16* lB0 = &ldsB[0][brow * LDSW + bcol];
  constexpr int BUFA = BS * LDSW, BUFB = BQ * LDSW;

  float best[2] = {-3.0e38f, -3.0e38f};
  int   bidx[2] = {0, 0};

  for (int sb = 0; sb < S_SPAN; sb += BS) {
    const int sBlock = sChunkBase + sb;
    if (tid < BS) invLds[tid] = invNorm[sBlock + tid];
    const _Float16* gA = sty + (size_t)(sBlock + arow) * D + acol;  // + k

    v8f acc[2][2] = {};

    // issue tile 0 into buffer 0
    {
      cp16(gA, lA0); cp16(gA + 8, lA0 + 8);
      cp16(gB, lB0); cp16(gB + 8, lB0 + 8);
      cp16(gB + 16, lB0 + 16); cp16(gB + 24, lB0 + 24);
    }

    for (int kt = 0; kt < KT; ++kt) {
      // prefetch next K-tile into buffer (kt+1)%3 (overlaps current compute)
      if (kt + 1 < KT) {
        const int nb = (kt + 1) % NBUF;
        const _Float16* nA = gA + (kt + 1) * BK;
        const _Float16* nB = gB + (kt + 1) * BK;
        _Float16* dA = lA0 + nb * BUFA;
        _Float16* dB = lB0 + nb * BUFB;
        cp16(nA, dA); cp16(nA + 8, dA + 8);
        cp16(nB, dB); cp16(nB + 8, dB + 8);
        cp16(nB + 16, dB + 16); cp16(nB + 24, dB + 24);
        WAIT_ASYNC(6);   // all but the 6 just-issued are complete -> tile kt ready
      } else {
        WAIT_ASYNC(0);
      }
      __syncthreads();   // single barrier/tile: WAR distance is 3 buffers

      const int cb = kt % NBUF;
      #pragma unroll
      for (int ks = 0; ks < 2; ++ks) {   // two K=32 sub-steps per tile
        v16h a[2], b[2];
        #pragma unroll
        for (int mi = 0; mi < 2; ++mi) {
          const _Float16* pa = &ldsA[cb][(wm * 32 + mi * 16 + r) * LDSW + ks * 32 + g * 8];
          v8h lo = *(const v8h*)pa;
          v8h hi = *(const v8h*)(pa + 16);
          a[mi] = __builtin_shufflevector(lo, hi, 0,1,2,3,4,5,6,7,8,9,10,11,12,13,14,15);
        }
        #pragma unroll
        for (int ni = 0; ni < 2; ++ni) {
          const _Float16* pb = &ldsB[cb][(wn * 32 + ni * 16 + r) * LDSW + ks * 32 + g * 8];
          v8h lo = *(const v8h*)pb;
          v8h hi = *(const v8h*)(pb + 16);
          b[ni] = __builtin_shufflevector(lo, hi, 0,1,2,3,4,5,6,7,8,9,10,11,12,13,14,15);
        }
        #pragma unroll
        for (int mi = 0; mi < 2; ++mi)
          #pragma unroll
          for (int ni = 0; ni < 2; ++ni)
            acc[mi][ni] = __builtin_amdgcn_wmma_f32_16x16x32_f16(
                false, a[mi], false, b[ni], (short)0, acc[mi][ni], false, false);
      }
    }

    // scale by 1/||style|| and fold into running argmax; C layout: M = v + 8*g
    #pragma unroll
    for (int mi = 0; mi < 2; ++mi) {
      const int mBase = wm * 32 + mi * 16 + 8 * g;
      #pragma unroll
      for (int ni = 0; ni < 2; ++ni) {
        #pragma unroll
        for (int v = 0; v < 8; ++v) {
          const int sLoc  = mBase + v;
          const int sGlob = sBlock + sLoc;
          const float sc  = acc[mi][ni][v] * invLds[sLoc];
          const bool upd  = (sGlob < NS) & (sc > best[ni]);   // branchless
          best[ni] = upd ? sc : best[ni];
          bidx[ni] = upd ? sGlob : bidx[ni];
        }
      }
    }
    __syncthreads();  // invLds + LDS buffers reused next sBlock
  }

  // merge lane <-> lane^16 (same column N, other 8 rows)
  #pragma unroll
  for (int ni = 0; ni < 2; ++ni) {
    float ob = __shfl_xor(best[ni], 16, 32);
    int   oi = __shfl_xor(bidx[ni], 16, 32);
    bool upd = (ob > best[ni]) | ((ob == best[ni]) & (oi < bidx[ni]));
    best[ni] = upd ? ob : best[ni];
    bidx[ni] = upd ? oi : bidx[ni];
  }
  if (lane < 16) {
    #pragma unroll
    for (int ni = 0; ni < 2; ++ni) {
      int col = wn * 32 + ni * 16 + lane;
      bestLds[wm][col] = best[ni];
      idxLds[wm][col]  = bidx[ni];
    }
  }
  __syncthreads();
  if (tid < BQ) {
    float b0 = bestLds[0][tid], b1 = bestLds[1][tid];
    int   i0 = idxLds[0][tid],  i1 = idxLds[1][tid];
    bool t = (b1 > b0) | ((b1 == b0) & (i1 < i0));
    pBest[(size_t)blockIdx.y * Q_PAD + qBase + tid] = t ? b1 : b0;
    pIdx [(size_t)blockIdx.y * Q_PAD + qBase + tid] = t ? i1 : i0;
  }
}

// ---------------- stage 3: final fp32 MSE against matched style row ----------------
__global__ void k3_final(const float* __restrict__ img, const float* __restrict__ target,
                         const int* __restrict__ styleSrc,
                         const float* __restrict__ pBest, const int* __restrict__ pIdx,
                         float* __restrict__ out) {
  __shared__ float red[8];
  __shared__ int sBestSh;
  const int q = blockIdx.x, tid = threadIdx.x;
  if (tid == 0) {
    float bb = -3.0e38f; int ii = 0;
    for (int ch = 0; ch < S_CHUNKS; ++ch) {
      float b = pBest[(size_t)ch * Q_PAD + q];
      int   i = pIdx [(size_t)ch * Q_PAD + q];
      if (b > bb || (b == bb && i < ii)) { bb = b; ii = i; }
    }
    sBestSh = ii;
  }
  __syncthreads();
  const int p  = styleSrc[sBestSh];
  const int pi = (p / NP) * 2, pj = (p % NP) * 2;
  const int qi = (q / NP) * 2, qj = (q % NP) * 2;
  float sum = 0.0f;
  for (int d = tid; d < D; d += 256) {
    int c = d / 9, r = d % 9, pr = r / 3, pc = r % 3;
    float sv = img   [c * (H * W) + (qi + pr) * W + (qj + pc)];
    float tv = target[c * (H * W) + (pi + pr) * W + (pj + pc)];
    float df = sv - tv;
    sum += df * df;
  }
  for (int off = 16; off; off >>= 1) sum += __shfl_down(sum, off, 32);
  if ((tid & 31) == 0) red[tid >> 5] = sum;
  __syncthreads();
  if (tid == 0) {
    float t = 0.0f;
    #pragma unroll
    for (int w = 0; w < 8; ++w) t += red[w];
    atomicAdd(out, t * (1.0f / ((float)D * (float)NQ)));
  }
}

// ---------------- host launcher ----------------
extern "C" void kernel_launch(void* const* d_in, const int* in_sizes, int n_in,
                              void* d_out, int out_size, void* d_ws, size_t ws_size,
                              hipStream_t stream) {
  const float* img = (const float*)d_in[0];   // d_in image
  const float* tgt = (const float*)d_in[1];   // target image
  float* out = (float*)d_out;

  char* ws = (char*)d_ws;
  _Float16* sty   = (_Float16*)(ws + OFF_STY);
  _Float16* syn   = (_Float16*)(ws + OFF_SYN);
  float*    invN  = (float*)(ws + OFF_INVN);
  int*      src   = (int*)(ws + OFF_SRC);
  float*    pBest = (float*)(ws + OFF_PBST);
  int*      pIdx  = (int*)(ws + OFF_PIDX);

  k0_setup<<<(NQ + 255) / 256, 256, 0, stream>>>(src, out);
  k1_style<<<S_PAD, 256, 0, stream>>>(tgt, src, sty, invN);
  k1_synth<<<Q_PAD, 256, 0, stream>>>(img, syn);
  dim3 g2(Q_PAD / BQ, S_CHUNKS);
  k2_gemm_argmax<<<g2, 256, 0, stream>>>(sty, syn, invN, pBest, pIdx);
  k3_final<<<NQ, 256, 0, stream>>>(img, tgt, src, pBest, pIdx, out);
}
